// DenseIDM_3083786518535
// MI455X (gfx1250) — matrix operand order repaired and checked by reading, not compiled, not run on hardware
//
#include <hip/hip_runtime.h>

typedef __attribute__((ext_vector_type(16))) _Float16 v16h;
typedef __attribute__((ext_vector_type(8)))  _Float16 v8h;
typedef __attribute__((ext_vector_type(4)))  _Float16 v4h;
typedef __attribute__((ext_vector_type(8)))  float    v8f;

// ---------------- weight fragment layout in d_ws (units: 512-half fragments) ---
#define FRAG_HALVES 512
#define FB_C1 0     // conv1: 1 frag
#define FB_C2 1     // conv2: nt(2) x kt(2)
#define FB_C3 5     // conv3: nt(4) x kt(4)
#define FB_PJ 21    // proj : nt(4) x kt(19)
#define FB_H1 97    // h1   : nt(8) x kt(5)
#define FB_H2 137   // h2   : nt(8) x kt(4)
#define FB_H3 169   // h3   : kt(4)
#define N_FRAGS 173

// per-wave LDS layout (units: halves).  h-stage buffers alias x2 (disjoint liveness).
#define OFF_XC   0        // conv1 padded input [16][9][9][4] (5184) / feats [16][608] (9728)
#define OFF_X1   9728     // conv2 padded input [16][6][6][16] (9216)
#define OFF_X2   18944    // conv3 padded input [16][4][4][32] (8192)
#define OFF_HBUF 18944    // f32 [16][128] (4096 halves)   -- aliases x2
#define OFF_SP   18944    // f32 [32][4]                   -- aliases hbuf (dead before h1)
#define OFF_A1   23040    // f16 [16][128] (2048)          -- aliases x2 tail
#define OFF_A2   25088    // f16 [16][128] (2048)
#define OFF_COMB 27136    // h1 input [16][160] (2560)
#define WAVE_HALVES 29696 // 59392 bytes / wave
#define WAVES_PER_BLOCK 2

// A-operand (16-bit, 16x32) per-lane K pattern: half j of lane ->
//   lanes 0-15 : {0..7, 16..23},  lanes 16-31 : {8..15, 24..31}
__device__ __forceinline__ int kpatA(int j, int hi) { return (j < 8 ? j : j + 8) + hi * 8; }

// ------------------------------------------------------------------ prep kernel
// Convert f32 weights to f16 WMMA B-fragments: B[k][n] = W[n][k'] with conv
// kernels reordered to (dy,dx,c) K-order. B layout: lane<16 holds K=kt*32+{0..7,16..23},
// lane>=16 holds K=kt*32+{8..15,24..31}, N = nt*16 + (lane&15).
__global__ void idm_prep_weights(_Float16* __restrict__ ws,
                                 const float* __restrict__ W1, const float* __restrict__ W2,
                                 const float* __restrict__ W3, const float* __restrict__ Wp,
                                 const float* __restrict__ Wh1, const float* __restrict__ Wh2,
                                 const float* __restrict__ Wh3) {
  int gid = blockIdx.x * blockDim.x + threadIdx.x;
  int fid = gid >> 5;
  int lane = gid & 31;
  if (fid >= N_FRAGS) return;
  int hi = lane >> 4;
  int n16 = lane & 15;

  int kind, nt, kt;
  if (fid < FB_C2)      { kind = 0; nt = 0; kt = 0; }
  else if (fid < FB_C3) { int f = fid - FB_C2; kind = 1; nt = f >> 1; kt = f & 1; }
  else if (fid < FB_PJ) { int f = fid - FB_C3; kind = 2; nt = f >> 2; kt = f & 3; }
  else if (fid < FB_H1) { int f = fid - FB_PJ; kind = 3; nt = f / 19; kt = f % 19; }
  else if (fid < FB_H2) { int f = fid - FB_H1; kind = 4; nt = f / 5;  kt = f % 5; }
  else if (fid < FB_H3) { int f = fid - FB_H2; kind = 5; nt = f >> 2; kt = f & 3; }
  else                  { int f = fid - FB_H3; kind = 6; nt = 0;      kt = f;     }

  int n = nt * 16 + n16;
  _Float16* dst = ws + (size_t)fid * FRAG_HALVES + lane * 16;
  for (int j = 0; j < 16; ++j) {
    int K = kt * 32 + kpatA(j, hi);  // B-operand table K layout (matches A)
    float w = 0.f;
    switch (kind) {
      case 0: if (K < 16 && (K & 3) < 3) w = W1[n * 12 + (K & 3) * 4 + (K >> 2)]; break;  // conv1, C padded 3->4
      case 1: if (K < 64)  w = W2[n * 64  + (K & 15) * 4 + (K >> 4)]; break;              // conv2
      case 2:              w = W3[n * 128 + (K & 31) * 4 + (K >> 5)]; break;              // conv3
      case 3: if (K < 578) w = Wp [n * 578 + K]; break;                                    // proj
      case 4: if (K < 132) w = Wh1[n * 132 + K]; break;                                    // h1
      case 5:              w = Wh2[n * 128 + K]; break;                                    // h2
      case 6: if (n < 7)   w = Wh3[n * 128 + K]; break;                                    // h3
    }
    dst[j] = (_Float16)w;
  }
}

// ------------------------------------------------------------------ main kernel
__global__ __launch_bounds__(WAVES_PER_BLOCK * 32)
void idm_forward(const int* __restrict__ frame, const int* __restrict__ ccol,
                 const int* __restrict__ cobj,
                 const float* __restrict__ b1, const float* __restrict__ b2,
                 const float* __restrict__ b3, const float* __restrict__ bp,
                 const float* __restrict__ bh1, const float* __restrict__ lng,
                 const float* __restrict__ lnb, const float* __restrict__ bh2,
                 const float* __restrict__ bh3, const _Float16* __restrict__ ws,
                 float* __restrict__ out, int Bn) {
  extern __shared__ __align__(16) char smem[];
  const int lane = threadIdx.x & 31;
  const int wave = threadIdx.x >> 5;
  const int tile = blockIdx.x * WAVES_PER_BLOCK + wave;
  const int base = tile * 16;
  if (base >= Bn) return;
  const int m = lane & 15;   // batch row (A-operand M, and D column index)
  const int hi = lane >> 4;  // which K/M half this lane covers

  _Float16* wl   = (_Float16*)smem + (size_t)wave * WAVE_HALVES;
  _Float16* xC   = wl + OFF_XC;
  _Float16* x1   = wl + OFF_X1;
  _Float16* x2   = wl + OFF_X2;
  _Float16* comb = wl + OFF_COMB;
  float*    hbuf = (float*)(wl + OFF_HBUF);
  _Float16* a1   = wl + OFF_A1;
  _Float16* a2   = wl + OFF_A2;
  float*    sp   = (float*)(wl + OFF_SP);

  // warm L2 with this tile's frame-1 data while frame 0 is processed
  {
    const char* pf = (const char*)(frame + (size_t)(Bn + base) * 147);
    __builtin_prefetch(pf + lane * 256, 0, 1);
    __builtin_prefetch(pf + 8192 + lane * 64, 0, 1);
  }

  auto zero_halves = [&](_Float16* p, int halves) {
    int* q = (int*)p;
    for (int i = lane; i < (halves >> 1); i += 32) q[i] = 0;
  };
  auto ldA = [&](const _Float16* p) -> v16h {  // p includes +hi*8; two 16B runs
    union { v16h v; v8h h[2]; } u;
    u.h[0] = *(const v8h*)p;
    u.h[1] = *(const v8h*)(p + 16);
    return u.v;
  };
  auto ldB = [&](int frag) -> v16h {
    return *(const v16h*)(ws + (size_t)frag * FRAG_HALVES + lane * 16);
  };
  auto mma = [&](v16h a, v16h b, v8f c) -> v8f {
    return __builtin_amdgcn_wmma_f32_16x16x32_f16(false, a, false, b, (short)0, c, false, false);
  };

  zero_halves(comb, 2560);

  for (int f = 0; f < 2; ++f) {
    const int* fp = frame + (size_t)(f * Bn + base) * 147;

    // ---- stage 1: padded conv1 input, channel-last [16][9][9][4] (c=3 stays 0)
    zero_halves(xC, 5184);
    for (int i = lane; i < 16 * 147; i += 32) {
      int b = i / 147, r = i % 147;
      int cell = r / 3, c = r % 3;
      int y = cell / 7, x = cell % 7;
      xC[b * 324 + (1 + y) * 36 + (1 + x) * 4 + c] = (_Float16)(float)fp[b * 147 + r];
    }

    // ---- conv1 (K=16 pad 32, N=16) + fused 2x2 maxpool -> x1 [16][6][6][16] (pad=1)
    // K-order (dy,dx,c), C padded to 4: K0..7  = 8 contiguous halves at row oy,
    //                                   K8..15 = 8 contiguous halves at row oy+1.
    // Lane split matches the A layout exactly -> two b64 LDS loads, upper half zero.
    zero_halves(x1, 9216);
    {
      v16h bf = ldB(FB_C1);
      float bias = b1[m];
      const v4h zq = {};
      for (int py = 0; py < 4; ++py)
        for (int px = 0; px < 4; ++px) {
          v8f mx = {};
          for (int s = 0; s < 4; ++s) {
            int oy = 2 * py + (s >> 1), ox = 2 * px + (s & 1);
            const _Float16* p = xC + m * 324 + (oy + hi) * 36 + ox * 4;  // 8B aligned
            union { v16h v; v4h q[4]; } u;
            u.q[0] = *(const v4h*)p;
            u.q[1] = *(const v4h*)(p + 4);
            u.q[2] = zq;
            u.q[3] = zq;
            v8f d = {};
            d = mma(u.v, bf, d);
            if (s == 0) mx = d;
            else {
#pragma unroll
              for (int r = 0; r < 8; ++r) mx[r] = fmaxf(mx[r], d[r]);
            }
          }
#pragma unroll
          for (int r = 0; r < 8; ++r) {
            float v = fmaxf(mx[r] + bias, 0.f);
            x1[(r + hi * 8) * 576 + ((1 + py) * 6 + (1 + px)) * 16 + m] = (_Float16)v;
          }
        }
    }

    // ---- conv2 (K=64, N=32) + maxpool (only pooled-covered positions) -> x2 [16][4][4][32]
    zero_halves(x2, 8192);
    for (int py = 0; py < 2; ++py)
      for (int px = 0; px < 2; ++px)
        for (int nt = 0; nt < 2; ++nt) {
          float bias = b2[nt * 16 + m];
          v8f mx = {};
          for (int s = 0; s < 4; ++s) {
            int oy = 2 * py + (s >> 1), ox = 2 * px + (s & 1);
            v8f d = {};
#pragma unroll
            for (int kt = 0; kt < 2; ++kt) {  // kt == dy; runs: dx=0 then dx=1
              const _Float16* row = x1 + m * 576 + ((oy + kt) * 6 + ox) * 16 + hi * 8;
              d = mma(ldA(row), ldB(FB_C2 + nt * 2 + kt), d);
            }
            if (s == 0) mx = d;
            else {
#pragma unroll
              for (int r = 0; r < 8; ++r) mx[r] = fmaxf(mx[r], d[r]);
            }
          }
#pragma unroll
          for (int r = 0; r < 8; ++r) {
            float v = fmaxf(mx[r] + bias, 0.f);
            x2[(r + hi * 8) * 512 + ((1 + py) * 4 + (1 + px)) * 32 + nt * 16 + m] = (_Float16)v;
          }
        }

    // ---- conv3 (K=128, N=64) -> feats [16][608] in NCHW-flatten order (c*9+oy*3+ox)
    zero_halves(xC, 9728);
    for (int oy = 0; oy < 3; ++oy)
      for (int ox = 0; ox < 3; ++ox)
        for (int nt = 0; nt < 4; ++nt) {
          v8f d = {};
#pragma unroll
          for (int kt = 0; kt < 4; ++kt) {  // kt == dy*2+dx, one 32-ch run each
            const _Float16* run = x2 + m * 512 + ((oy + (kt >> 1)) * 4 + (ox + (kt & 1))) * 32 + hi * 8;
            d = mma(ldA(run), ldB(FB_C3 + nt * 4 + kt), d);
          }
          float bias = b3[nt * 16 + m];
#pragma unroll
          for (int r = 0; r < 8; ++r) {
            float v = fmaxf(d[r] + bias, 0.f);
            xC[(r + hi * 8) * 608 + (nt * 16 + m) * 9 + oy * 3 + ox] = (_Float16)v;
          }
        }
    if (lane < 16) {
      xC[lane * 608 + 576] = (_Float16)(float)ccol[f * Bn + base + lane];
      xC[lane * 608 + 577] = (_Float16)(float)cobj[f * Bn + base + lane];
    }

    // ---- proj (K=578 pad 608, N=64) + relu -> comb columns [f*64, f*64+64)
    for (int nt = 0; nt < 4; ++nt) {
      v8f d = {};
      for (int kt = 0; kt < 19; ++kt)
        d = mma(ldA(xC + m * 608 + kt * 32 + hi * 8), ldB(FB_PJ + nt * 19 + kt), d);
      float bias = bp[nt * 16 + m];
#pragma unroll
      for (int r = 0; r < 8; ++r)
        comb[(r + hi * 8) * 160 + f * 64 + nt * 16 + m] = (_Float16)fmaxf(d[r] + bias, 0.f);
    }
  }

  // ---- dir/pos features: lane = f*16 + b scans its frame
  {
    const int* fp = frame + (size_t)(hi * Bn + base + m) * 147;
    int idx = 0, found = 0;
    for (int i = 0; i < 49; ++i) {
      int v = fp[i * 3];
      if (!found && v == 10) { found = 1; idx = i; }
    }
    int d = found ? (fp[idx * 3 + 2] & 3) : 0;
    float py = found ? (float)(idx / 7) * (1.f / 6.f) : 0.5f;
    float px = found ? (float)(idx % 7) * (1.f / 6.f) : 0.5f;
    sp[lane * 4 + 0] = (float)d;
    sp[lane * 4 + 1] = py;
    sp[lane * 4 + 2] = px;
  }
  if (lane < 16) {
    int cd = (int)sp[lane * 4 + 0], nd = (int)sp[(lane + 16) * 4 + 0];
    int delta = (nd - cd + 4) & 3;
    float ang = (float)delta * (6.28318f * 0.25f);
    comb[lane * 160 + 128] = (_Float16)__sinf(ang);
    comb[lane * 160 + 129] = (_Float16)__cosf(ang);
    comb[lane * 160 + 130] = (_Float16)(sp[(lane + 16) * 4 + 1] - sp[lane * 4 + 1]);
    comb[lane * 160 + 131] = (_Float16)(sp[(lane + 16) * 4 + 2] - sp[lane * 4 + 2]);
  }

  // ---- h1 (K=132 pad 160, N=128) -> hbuf f32 (pre-LN)
  for (int nt = 0; nt < 8; ++nt) {
    v8f d = {};
#pragma unroll
    for (int kt = 0; kt < 5; ++kt)
      d = mma(ldA(comb + m * 160 + kt * 32 + hi * 8), ldB(FB_H1 + nt * 5 + kt), d);
    float bias = bh1[nt * 16 + m];
#pragma unroll
    for (int r = 0; r < 8; ++r) hbuf[(r + hi * 8) * 128 + nt * 16 + m] = d[r] + bias;
  }
  // ---- layernorm + relu (one row per lane 0..15)
  if (lane < 16) {
    const float* hr = hbuf + lane * 128;
    float s = 0.f, ss = 0.f;
    for (int j = 0; j < 128; ++j) { float v = hr[j]; s += v; ss += v * v; }
    float mu = s * (1.f / 128.f);
    float var = ss * (1.f / 128.f) - mu * mu;
    float rstd = rsqrtf(var + 1e-5f);
    for (int j = 0; j < 128; ++j) {
      float v = (hr[j] - mu) * rstd * lng[j] + lnb[j];
      a1[lane * 128 + j] = (_Float16)fmaxf(v, 0.f);
    }
  }
  // ---- h2 (K=128, N=128) + relu
  for (int nt = 0; nt < 8; ++nt) {
    v8f d = {};
#pragma unroll
    for (int kt = 0; kt < 4; ++kt)
      d = mma(ldA(a1 + m * 128 + kt * 32 + hi * 8), ldB(FB_H2 + nt * 4 + kt), d);
    float bias = bh2[nt * 16 + m];
#pragma unroll
    for (int r = 0; r < 8; ++r)
      a2[(r + hi * 8) * 128 + nt * 16 + m] = (_Float16)fmaxf(d[r] + bias, 0.f);
  }
  // ---- h3 (K=128, N=7 pad 16) -> global out
  {
    v8f d = {};
#pragma unroll
    for (int kt = 0; kt < 4; ++kt)
      d = mma(ldA(a2 + m * 128 + kt * 32 + hi * 8), ldB(FB_H3 + kt), d);
    if (m < 7) {
      float bias = bh3[m];
#pragma unroll
      for (int r = 0; r < 8; ++r)
        out[(size_t)(base + r + hi * 8) * 7 + m] = d[r] + bias;
    }
  }
}

// ------------------------------------------------------------------ launcher
extern "C" void kernel_launch(void* const* d_in, const int* in_sizes, int n_in,
                              void* d_out, int out_size, void* d_ws, size_t ws_size,
                              hipStream_t stream) {
  const int*   frame = (const int*)d_in[0];
  const int*   ccol  = (const int*)d_in[1];
  const int*   cobj  = (const int*)d_in[2];
  const float* W1 = (const float*)d_in[3];  const float* b1  = (const float*)d_in[4];
  const float* W2 = (const float*)d_in[5];  const float* b2  = (const float*)d_in[6];
  const float* W3 = (const float*)d_in[7];  const float* b3  = (const float*)d_in[8];
  const float* Wp = (const float*)d_in[9];  const float* bp  = (const float*)d_in[10];
  const float* Wh1 = (const float*)d_in[11]; const float* bh1 = (const float*)d_in[12];
  const float* lng = (const float*)d_in[13]; const float* lnb = (const float*)d_in[14];
  const float* Wh2 = (const float*)d_in[15]; const float* bh2 = (const float*)d_in[16];
  const float* Wh3 = (const float*)d_in[17]; const float* bh3 = (const float*)d_in[18];
  _Float16* ws = (_Float16*)d_ws;
  float* outp = (float*)d_out;

  int Bn = in_sizes[0] / 294;  // [2,B,7,7,3]

  idm_prep_weights<<<(N_FRAGS * 32 + 63) / 64, 64, 0, stream>>>(ws, W1, W2, W3, Wp, Wh1, Wh2, Wh3);

  int tiles = Bn / 16;
  int blocks = (tiles + WAVES_PER_BLOCK - 1) / WAVES_PER_BLOCK;
  size_t smem = (size_t)WAVES_PER_BLOCK * WAVE_HALVES * sizeof(_Float16);
  idm_forward<<<blocks, WAVES_PER_BLOCK * 32, smem, stream>>>(
      frame, ccol, cobj, b1, b2, b3, bp, bh1, lng, lnb, bh2, bh3, ws, outp, Bn);
}
